// WoquColumnParallelLinear_32341103739318
// MI455X (gfx1250) — matrix operand order repaired and checked by reading, not compiled, not run on hardware
//
#include <hip/hip_runtime.h>

// ---- problem geometry (fixed by the reference) ----
#define TOKENS 4096
#define IN_F   4096
#define OUT_F  12288
#define GROUP  128
#define QROWS  (OUT_F / 8)     // 1536 packed weight rows

// ---- tiling ----
#define BM  128           // M rows per workgroup
#define BN  128           // N cols per workgroup
#define KS  64            // K per double-buffered stage
#define LDA 72            // padded LDS row stride in f16 elems (16B aligned, 36-dword stride)
#define NKT (IN_F / KS)

typedef __attribute__((ext_vector_type(16))) _Float16 v16h;
typedef __attribute__((ext_vector_type(2)))  _Float16 v2h;
typedef __attribute__((ext_vector_type(8)))  float    v8f;
typedef __attribute__((ext_vector_type(4)))  unsigned v4u;
typedef __attribute__((ext_vector_type(8)))  int      v8i;
typedef __attribute__((ext_vector_type(4)))  int      v4i;

union Frag { uint4 q[2]; v16h f; };

// pack two f32 -> packed f16 dword (v_cvt_pk_rtz_f16_f32, single instruction)
__device__ __forceinline__ unsigned pk2(float lo, float hi) {
  return __builtin_bit_cast(unsigned, __builtin_amdgcn_cvt_pkrtz(lo, hi));
}

// ---- Tensor Data Mover: DMA one 16-row x 64-dword qweight tile into LDS ----
__device__ __forceinline__ void tdm_load_qtile(const int* gptr, unsigned lds_byte_off) {
  const unsigned long long ga = (unsigned long long)(uintptr_t)gptr;
  v4u g0;
  g0.x = 1u;                                                  // count=1, user-mode D#
  g0.y = lds_byte_off;                                        // lds_addr [63:32]
  g0.z = (unsigned)(ga & 0xFFFFFFFFu);                        // global_addr[31:0]
  g0.w = (unsigned)((ga >> 32) & 0x01FFFFFFu) | (2u << 30);   // global_addr[56:32] | type=2
  v8i g1;
  g1[0] = 0x00020000;                                         // data_size=4B, no multicast/pad
  g1[1] = (int)((unsigned)(IN_F & 0xFFFF) << 16);             // tensor_dim0[15:0]  @ [63:48]
  g1[2] = (int)(((IN_F >> 16) & 0xFFFF) |
                ((QROWS & 0xFFFF) << 16));                    // dim0 hi | tensor_dim1 lo
  g1[3] = (int)(((QROWS >> 16) & 0xFFFF) | (64u << 16));      // dim1 hi | tile_dim0=64
  g1[4] = 16;                                                 // tile_dim1=16, tile_dim2=0
  g1[5] = IN_F;                                               // tensor_dim0_stride[31:0]
  g1[6] = 0;                                                  // stride hi | dim1_stride lo
  g1[7] = 0;
  const v4i z4 = {0, 0, 0, 0};
#if defined(__clang_major__) && __clang_major__ >= 23
  const v8i z8 = {0, 0, 0, 0, 0, 0, 0, 0};
  __builtin_amdgcn_tensor_load_to_lds(g0, g1, z4, z4, z8, 0);
#else
  __builtin_amdgcn_tensor_load_to_lds(g0, g1, z4, z4, 0);
#endif
}

__global__ __launch_bounds__(256, 1)
void WoquColumnParallelLinear_kernel(const float* __restrict__ X,
                                     const int*   __restrict__ qweight,
                                     const float* __restrict__ scale,
                                     const float* __restrict__ bias,
                                     float* __restrict__ out) {
  __shared__ unsigned short Xs[2][BM * LDA];             // f16 activations
  __shared__ unsigned short Ws[2][BN * LDA];             // f16 dequantized weights, [n][k]
  __shared__ float          Ss[BN * (IN_F / GROUP)];     // per-block scales [128][32]
  __shared__ unsigned       Qs[2][16 * 64];              // raw packed weight tiles (TDM dest)

  const int tid  = threadIdx.x;
  const int lane = tid & 31;
  const int wave = tid >> 5;
  const int n_wg = blockIdx.x * BN;
  const int m_wg = blockIdx.y * BM;

  // generic LDS pointers keep the byte offset in addr[31:0] (flat aperture rule)
  const unsigned qs_off0 = (unsigned)(uintptr_t)(void*)&Qs[0][0];
  const unsigned qs_off1 = (unsigned)(uintptr_t)(void*)&Qs[1][0];
  const int* qtile_base = qweight + (size_t)(n_wg / 8) * IN_F;

  auto issue_tdm = [&](int kt) {        // wave 0 only; EXEC ignored by TDM
    tdm_load_qtile(qtile_base + kt * KS, (kt & 1) ? qs_off1 : qs_off0);
  };

  // ---- stage this block's scales into LDS (once) ----
  for (int idx = tid; idx < BN * (IN_F / GROUP); idx += 256) {
    const int nl = idx >> 5, g = idx & 31;
    Ss[idx] = scale[(size_t)(n_wg + nl) * (IN_F / GROUP) + g];
  }

  // ---- loader geometry ----
  const int  xr_row   = tid & 127;        // X tile row 0..127
  const int  xr_kh    = tid >> 7;         // 0/1 -> k offset 0/32
  const int  wl_o8    = tid >> 3;         // packed-weight row 0..15 (threads < 128)
  const int  wl_kc    = (tid & 7) * 8;    // k_local 0,8,...,56
  const bool is_wload = (tid < 128);

  float4 xr[8];

  auto load_x = [&](int kt) {
    const int kb = kt * KS;
    const float* xp = X + (size_t)(m_wg + xr_row) * IN_F + kb + xr_kh * 32;
#pragma unroll
    for (int j = 0; j < 8; ++j) xr[j] = ((const float4*)xp)[j];
    if (kt + 1 < NKT)                     // speculative prefetch of stage kt+1
      __builtin_prefetch(X + (size_t)(m_wg + xr_row) * IN_F + kb + KS + xr_kh * 32, 0, 0);
  };

  // f16 magic constants: 0x6400|t == (1024+t) exactly; 1032 == 0x6408
  const v2h koff = __builtin_bit_cast(v2h, 0x64086408u);

  auto store_lds = [&](int buf, int kt) {
    // X: 32 fp32 -> 32 f16 via 16x v_cvt_pk_rtz_f16_f32 -> 4x ds_store_b128
    unsigned short* xd = &Xs[buf][xr_row * LDA + xr_kh * 32];
#pragma unroll
    for (int j = 0; j < 4; ++j) {
      const float4 a = xr[2 * j], b = xr[2 * j + 1];
      uint4 p;
      p.x = pk2(a.x, a.y);  p.y = pk2(a.z, a.w);
      p.z = pk2(b.x, b.y);  p.w = pk2(b.z, b.w);
      *(uint4*)&xd[j * 8] = p;
    }
    // W: raw packed words from the TDM-staged LDS tile -> packed-f16 dequant
    if (is_wload) {
      const int g = (kt * KS) >> 7;       // quant group (KS divides GROUP)
      const unsigned* qsrc = &Qs[kt & 1][wl_o8 * 64 + wl_kc];
      const uint4 w0 = *(const uint4*)qsrc;
      const uint4 w1 = *(const uint4*)(qsrc + 4);
      unsigned wx[8] = {w0.x, w0.y, w0.z, w0.w, w1.x, w1.y, w1.z, w1.w};
#pragma unroll
      for (int q = 0; q < 8; ++q) wx[q] ^= 0x88888888u;   // nibble -> t = nib^8
#pragma unroll
      for (int j = 0; j < 8; ++j) {       // j = nibble index = out-feature offset
        const int      nl = wl_o8 * 8 + j;
        const float    s  = Ss[nl * (IN_F / GROUP) + g];
        const _Float16 hs = (_Float16)s;
        const v2h      sp = {hs, hs};
        uint4 p;
        unsigned o[4];
#pragma unroll
        for (int pr = 0; pr < 4; ++pr) {  // pair of consecutive k
          const unsigned a0 = wx[2 * pr]     >> (4 * j);
          const unsigned a1 = wx[2 * pr + 1] >> (4 * j);
          const unsigned u  = 0x64006400u | (a0 & 0xFu) | ((a1 & 0xFu) << 16);
          v2h h = __builtin_bit_cast(v2h, u);
          h = (h - koff) * sp;            // exact (1024+t)-1032 = t-8, then * s
          o[pr] = __builtin_bit_cast(unsigned, h);
        }
        p.x = o[0]; p.y = o[1]; p.z = o[2]; p.w = o[3];
        *(uint4*)&Ws[buf][nl * LDA + wl_kc] = p;
      }
    }
  };

  // ---- per-wave compute geometry: 2(M) x 4(N) wave grid, 64x32 region per wave ----
  const int wm   = wave >> 2;             // 0..1
  const int wn   = wave & 3;              // 0..3
  const int lrow = lane & 15;
  const int asel = (lane >> 4) * 8;       // A frag: hi half-wave starts at K=8
  const int bsel = (lane >> 4) * 8;       // B frag: hi half-wave starts at K=8

  v8f acc[4][2];
  {
    const float b0 = bias[n_wg + wn * 32 + lrow];
    const float b1 = bias[n_wg + wn * 32 + 16 + lrow];
#pragma unroll
    for (int i = 0; i < 4; ++i)
#pragma unroll
      for (int r = 0; r < 8; ++r) { acc[i][0][r] = b0; acc[i][1][r] = b1; }
  }

  auto mma_step = [&](int buf) {
#pragma unroll
    for (int kc = 0; kc < KS; kc += 32) {
      Frag b[2];
#pragma unroll
      for (int j = 0; j < 2; ++j) {
        const unsigned short* bp = &Ws[buf][(wn * 32 + j * 16 + lrow) * LDA + kc + bsel];
        b[j].q[0] = *(const uint4*)bp;
        b[j].q[1] = *(const uint4*)(bp + 16);
      }
#pragma unroll
      for (int i = 0; i < 4; ++i) {
        Frag a;
        const unsigned short* ap = &Xs[buf][(wm * 64 + i * 16 + lrow) * LDA + kc + asel];
        a.q[0] = *(const uint4*)ap;
        a.q[1] = *(const uint4*)(ap + 16);
#pragma unroll
        for (int j = 0; j < 2; ++j)
          acc[i][j] = __builtin_amdgcn_wmma_f32_16x16x32_f16(
              false, a.f, false, b[j].f, (short)0, acc[i][j], false, false);
      }
    }
  };

  // ---- pipeline: double-buffered stages, TDM weight DMA at distance 2 ----
  if (wave == 0) {
    issue_tdm(0);
    if (NKT > 1) issue_tdm(1);
  }
  load_x(0);
  if (wave == 0) __builtin_amdgcn_s_wait_tensorcnt((short)(NKT > 1 ? 1 : 0));
  __syncthreads();                        // Ss + Qs[0] ready
  store_lds(0, 0);
  __syncthreads();
  for (int kt = 0; kt < NKT; ++kt) {
    const int buf = kt & 1;
    if (wave == 0 && kt + 2 < NKT) issue_tdm(kt + 2);
    if (kt + 1 < NKT) load_x(kt + 1);
    mma_step(buf);
    if (kt + 1 < NKT) {
      if (wave == 0) {                    // TDM(kt+1) landed; TDM(kt+2) may still fly
        if (kt + 2 < NKT) __builtin_amdgcn_s_wait_tensorcnt((short)1);
        else              __builtin_amdgcn_s_wait_tensorcnt((short)0);
      }
      __syncthreads();                    // publish Qs[(kt+1)&1] to all waves
      store_lds(buf ^ 1, kt + 1);
    }
    __syncthreads();
  }

  // ---- epilogue: C/D layout -> global (half-wave rows are 64B contiguous) ----
  const int row_off = (lane >> 4) * 8;
#pragma unroll
  for (int i = 0; i < 4; ++i)
#pragma unroll
    for (int j = 0; j < 2; ++j) {
      const int col = n_wg + wn * 32 + j * 16 + lrow;
      float* op = out + (size_t)(m_wg + wm * 64 + i * 16 + row_off) * OUT_F + col;
#pragma unroll
      for (int r = 0; r < 8; ++r) op[(size_t)r * OUT_F] = acc[i][j][r];
    }
}

extern "C" void kernel_launch(void* const* d_in, const int* in_sizes, int n_in,
                              void* d_out, int out_size, void* d_ws, size_t ws_size,
                              hipStream_t stream) {
  const float* X  = (const float*)d_in[0];
  const int*   qw = (const int*)d_in[1];
  const float* sc = (const float*)d_in[2];
  const float* bs = (const float*)d_in[3];
  float*       o  = (float*)d_out;
  dim3 grid(OUT_F / BN, TOKENS / BM);     // 96 x 32 workgroups
  hipLaunchKernelGGL(WoquColumnParallelLinear_kernel, grid, dim3(256), 0, stream,
                     X, qw, sc, bs, o);
}
